// CrossLayerAggregation_34273839022117
// MI455X (gfx1250) — compile-verified, matrix operand order checked by reading, not run on hardware
//
#include <hip/hip_runtime.h>

// CrossLayerAggregation for MI455X (gfx1250).
// bf16 WMMA GEMMs, bf16 transposed [L][C] intermediates (half traffic,
// 16B packed epilogue stores, copy-only B staging), async global->LDS staging,
// fused fold/conv/pool epilogues. B=8, C=256, LUP=16384, LDOWN=32768.

#define GK    256   // contraction (channels)
#define KS    32    // WMMA K per step
#define PITCH 40    // LDS row pitch in bf16 elems (80B: 16B-aligned, conflict-free frags)

typedef __attribute__((ext_vector_type(16))) __bf16 v16bf;
typedef __attribute__((ext_vector_type(8)))  float  v8f;

__device__ __forceinline__ unsigned short f2bf(float f) {
  union { __bf16 h; unsigned short u; } c;
  c.h = (__bf16)f;          // RNE convert; clang can pack as v_cvt_pk_bf16_f32
  return c.u;
}

// ---- gfx1250 async global->LDS copy (ASYNCcnt path), with safe fallback ----
#if __has_builtin(__builtin_amdgcn_global_load_async_to_lds_b128) && \
    __has_builtin(__builtin_amdgcn_s_wait_asynccnt)
#define HAVE_ASYNC 1
#else
#define HAVE_ASYNC 0
#endif

#if HAVE_ASYNC
typedef int v4i_async __attribute__((vector_size(16)));
typedef __attribute__((address_space(1))) v4i_async* g_v4i_p;   // global int4*
typedef __attribute__((address_space(3))) v4i_async* l_v4i_p;   // LDS int4*
#endif

__device__ __forceinline__ void copy16_g2l(const void* g, void* l) {
#if HAVE_ASYNC
  __builtin_amdgcn_global_load_async_to_lds_b128(
      (g_v4i_p)(unsigned long long)g,
      (l_v4i_p)(unsigned int)(unsigned long long)l,
      0, 0);
#else
  *(uint4*)l = *(const uint4*)g;
#endif
}
__device__ __forceinline__ void async_drain() {
#if HAVE_ASYNC
  __builtin_amdgcn_s_wait_asynccnt(0);
#endif
}

union FragU { v16bf v; unsigned int u[8]; };

// 16x32 bf16 WMMA fragment from LDS stored [row][k] (K-minor), PITCH elems/row.
// ISA 7.12.2: lane m(0-15) row M=m; VGPR i holds K pair {2i | 16+2(i-4)} + 8*(lane>=16).
__device__ __forceinline__ v16bf load_frag(const unsigned short* lds, int row0, int lane) {
  FragU f;
  int m = lane & 15, h = lane >> 4;
  const unsigned short* p = lds + (row0 + m) * PITCH + 8 * h;
#pragma unroll
  for (int i = 0; i < 8; ++i) {
    int kb = (i < 4) ? (2 * i) : (16 + 2 * (i - 4));
    f.u[i] = *(const unsigned int*)(p + kb);
  }
  return f.v;
}

// ---------------------------------------------------------------------------
// Y = W . Xcol (+bias*scale) (+src) (+maxpool3_s2(pool))
// Block tile 128(M=o) x 128(N=l), 8 waves of 64x32, K = GK in steps of 32.
// BMODE 0: X f32 N-layout [k][l], col l            (conv1x1 on raw input)
// BMODE 1: X f32 N-layout, cols 2l & 2l+1 summed   (fold2 = interp adjoint)
// BMODE 2: X bf16 T-layout [l][k], row 2l+tap-1    (conv3_s2, 3 taps, zero pad)
// BMODE 3: X bf16 T-layout [l][k], row l           (conv1x1 on intermediate)
// OMODE 0: Y f32 N-layout [o][l]
// OMODE 1: Y bf16 T-layout [l][o] (16B packed stores: 8 consecutive channels)
// ---------------------------------------------------------------------------
template<int BMODE, int OMODE, bool ADD_SRC, bool ADD_POOL>
__global__ __launch_bounds__(256)
void gemm_k256(const float* __restrict__ W, int w_ld, int w_sk, long long w_batch,
               const void* __restrict__ X, long long x_batch, int x_row, int x_coff, int x_len,
               const float* __restrict__ bias, float bias_scale,
               const float* __restrict__ src, long long src_batch,
               const float* __restrict__ pool, long long pool_batch,
               void* __restrict__ Y, long long y_batch, int y_row, int Lout)
{
  constexpr int NTAPS = (BMODE == 2) ? 3 : 1;
  __shared__ unsigned short Alds[128 * PITCH];   // [o][k] bf16
  __shared__ unsigned short Blds[128 * PITCH];   // [l][k] bf16 (K-minor)
  const int t    = threadIdx.x;
  const int lane = t & 31;
  const int wid  = t >> 5;
  const int bz    = blockIdx.z;
  const int n_blk = blockIdx.x * 128;
  const int m_blk = blockIdx.y * 128;
  const int wm = (wid >> 2) * 64;   // 0 or 64
  const int wn = (wid & 3) * 32;    // 0,32,64,96

  v8f acc[4][2];
#pragma unroll
  for (int i = 0; i < 4; ++i)
#pragma unroll
    for (int j = 0; j < 2; ++j) acc[i][j] = v8f{0.f,0.f,0.f,0.f,0.f,0.f,0.f,0.f};

  const float* Wb = W + (long long)bz * w_batch;

  for (int tap = 0; tap < NTAPS; ++tap) {
    for (int ks = 0; ks < GK / KS; ++ks) {
      { // stage A: 128 rows (o) x 32 k, f32 -> bf16
        int o  = t >> 1;
        int k0 = (t & 1) * 16;
        const float* wp = Wb + (long long)(m_blk + o) * w_ld + (ks * KS + k0) * w_sk + tap;
        unsigned short* ap = Alds + o * PITCH + k0;
#pragma unroll
        for (int j = 0; j < 16; ++j) ap[j] = f2bf(wp[j * w_sk]);
      }
      if (BMODE <= 1) { // stage B from f32 N-layout: 32 k rows x 128 cols, transpose
        const float* Xb = (const float*)X + bz * x_batch;
        int k  = t >> 3;
        int c0 = (t & 7) * 16;
        const float* xp = Xb + (long long)(ks * KS + k) * x_row;
#pragma unroll
        for (int j = 0; j < 16; ++j) {
          int g = n_blk + c0 + j;
          float v = (BMODE == 0) ? xp[g] : (xp[2 * g] + xp[2 * g + 1]);
          Blds[(c0 + j) * PITCH + k] = f2bf(v);
        }
      } else { // stage B from bf16 T-layout: contiguous 16B chunk copies
        const unsigned short* Xb = (const unsigned short*)X + bz * x_batch;
#pragma unroll
        for (int i = 0; i < 2; ++i) {
          int id = t * 2 + i;
          int g  = id >> 2;           // 0..127 (tile row)
          int s8 = (id & 3) * 8;      // k sub-offset
          unsigned short* dst = &Blds[g * PITCH + s8];
          if (BMODE == 3) {
            const unsigned short* sp =
                Xb + (long long)(n_blk + g) * x_row + x_coff + ks * KS + s8;
            copy16_g2l(sp, dst);
          } else { // BMODE 2: strided conv taps with zero padding
            int row2 = 2 * (n_blk + g) + tap - 1;
            if (row2 >= 0 && row2 < x_len) {
              const unsigned short* sp =
                  Xb + (long long)row2 * x_row + x_coff + ks * KS + s8;
              *(uint4*)dst = *(const uint4*)sp;
            } else {
              *(uint4*)dst = uint4{0u,0u,0u,0u};
            }
          }
        }
      }
      if (BMODE == 3) async_drain();
      __syncthreads();

      v16bf bfr[2];
#pragma unroll
      for (int j = 0; j < 2; ++j) bfr[j] = load_frag(Blds, wn + j * 16, lane);
#pragma unroll
      for (int i = 0; i < 4; ++i) {
        v16bf afr = load_frag(Alds, wm + i * 16, lane);
#pragma unroll
        for (int j = 0; j < 2; ++j)
          acc[i][j] = __builtin_amdgcn_wmma_f32_16x16x32_bf16(
              false, afr, false, bfr[j], (short)0, acc[i][j], false, false);
      }
      __syncthreads();
    }
  }

  // Epilogue. C tile: lane n(0-15) col N=n; VGPR r row M=r+8*(lane>=16).
  int nl = lane & 15, h = lane >> 4;
#pragma unroll
  for (int i = 0; i < 4; ++i) {
#pragma unroll
    for (int j = 0; j < 2; ++j) {
      int col = n_blk + wn + j * 16 + nl;
      int o0  = m_blk + wm + i * 16 + h * 8;
      if (OMODE == 0) {
        float* Yf = (float*)Y + (long long)bz * y_batch;
#pragma unroll
        for (int r = 0; r < 8; ++r) {
          int o = o0 + r;
          float v = acc[i][j][r];
          if (bias) v += bias[o] * bias_scale;
          if (ADD_SRC) v += src[(long long)bz * src_batch + (long long)o * Lout + col];
          Yf[(long long)o * Lout + col] = v;
        }
      } else {
        unsigned short* Yt = (unsigned short*)Y + (long long)bz * y_batch;
        union { uint4 q; unsigned short s[8]; } pk;
#pragma unroll
        for (int r = 0; r < 8; ++r) {
          int o = o0 + r;
          float v = acc[i][j][r];
          if (bias) v += bias[o] * bias_scale;
          if (ADD_SRC) v += src[(long long)bz * src_batch + (long long)o * Lout + col];
          if (ADD_POOL) {
            const float* pp = pool + (long long)bz * pool_batch + (long long)o * x_len;
            float p  = fmaxf(pp[2 * col], pp[2 * col + 1]);
            if (col > 0) p = fmaxf(p, pp[2 * col - 1]);
            v += p;
          }
          pk.s[r] = f2bf(v);
        }
        *(uint4*)(Yt + (long long)col * y_row + o0) = pk.q;
      }
    }
  }
}

// ---------------------------------------------------------------------------
// S[b,c,d] += scale * sum_{j in chunk} QT[b,j,c] * KT[b,j,d]  (split-K atomics)
// QT/KT are bf16 T-layout [j][channels]. Block: 64x64 tile, j-chunk 2048.
// ---------------------------------------------------------------------------
__global__ __launch_bounds__(256)
void gemm_nt_atomic(const unsigned short* __restrict__ Q, long long q_batch, int q_row,
                    const unsigned short* __restrict__ K, long long k_batch, int k_row,
                    int k_coff, float scale, float* __restrict__ S)
{
  __shared__ unsigned short Qlds[64 * PITCH];   // [c][j]
  __shared__ unsigned short Klds[64 * PITCH];   // [d][j]
  int t = threadIdx.x, lane = t & 31, wid = t >> 5;
  int bz = blockIdx.z;
  int tm = (blockIdx.x >> 2) * 64;
  int tn = (blockIdx.x & 3) * 64;
  int j0 = blockIdx.y * 2048;
  int wm = (wid >> 2) * 32, wn = (wid & 3) * 16;
  v8f acc0 = v8f{0.f,0.f,0.f,0.f,0.f,0.f,0.f,0.f};
  v8f acc1 = acc0;
  const unsigned short* Qb = Q + (long long)bz * q_batch;
  const unsigned short* Kb = K + (long long)bz * k_batch;

  for (int ks = 0; ks < 2048 / KS; ++ks) {
    int j  = t >> 3;          // 0..31 (contraction row)
    int c0 = (t & 7) * 8;     // 0..56 (channel sub-range)
    long long jg = j0 + ks * KS + j;
    union { uint4 v; unsigned short s[8]; } qu, ku;
    qu.v = *(const uint4*)(Qb + jg * q_row + tm + c0);
    ku.v = *(const uint4*)(Kb + jg * k_row + k_coff + tn + c0);
#pragma unroll
    for (int i = 0; i < 8; ++i) {     // transpose into [c][j]
      Qlds[(c0 + i) * PITCH + j] = qu.s[i];
      Klds[(c0 + i) * PITCH + j] = ku.s[i];
    }
    __syncthreads();
    v16bf bfr = load_frag(Klds, wn, lane);
    v16bf a0  = load_frag(Qlds, wm, lane);
    v16bf a1  = load_frag(Qlds, wm + 16, lane);
    acc0 = __builtin_amdgcn_wmma_f32_16x16x32_bf16(false, a0, false, bfr, (short)0, acc0, false, false);
    acc1 = __builtin_amdgcn_wmma_f32_16x16x32_bf16(false, a1, false, bfr, (short)0, acc1, false, false);
    __syncthreads();
  }

  int nl = lane & 15, h = lane >> 4;
  int d  = tn + wn + nl;
#pragma unroll
  for (int r = 0; r < 8; ++r) {
    int c0r = tm + wm + h * 8 + r;
    atomicAdd(&S[((long long)bz * 256 + c0r) * 256 + d], acc0[r] * scale);
    atomicAdd(&S[((long long)bz * 256 + c0r + 16) * 256 + d], acc1[r] * scale);
  }
}

// In-place softmax over rows of 256 (one wave per row).
__global__ __launch_bounds__(256)
void softmax_rows(float* __restrict__ S, int nrows)
{
  int row  = blockIdx.x * 8 + (threadIdx.x >> 5);
  int lane = threadIdx.x & 31;
  if (row >= nrows) return;
  float* p = S + (long long)row * 256;
  float vals[8], m = -3.402823466e38f;
#pragma unroll
  for (int i = 0; i < 8; ++i) { vals[i] = p[lane + 32 * i]; m = fmaxf(m, vals[i]); }
#pragma unroll
  for (int off = 16; off >= 1; off >>= 1) m = fmaxf(m, __shfl_xor(m, off, 32));
  float s = 0.f;
#pragma unroll
  for (int i = 0; i < 8; ++i) { vals[i] = __expf(vals[i] - m); s += vals[i]; }
#pragma unroll
  for (int off = 16; off >= 1; off >>= 1) s += __shfl_xor(s, off, 32);
  float inv = 1.f / s;
#pragma unroll
  for (int i = 0; i < 8; ++i) p[lane + 32 * i] = vals[i] * inv;
}

// down_out[.,2j]=down[.,2j]+Ot[.,j]; down_out[.,2j+1]=down[.,2j+1]+Ot[.,j]
__global__ __launch_bounds__(256)
void dup_add(const float* __restrict__ down, const float* __restrict__ Ot,
             float* __restrict__ out, long long n)
{
  long long i = (long long)blockIdx.x * blockDim.x + threadIdx.x;
  if (i >= n) return;
  float o  = Ot[i];
  float2 d = ((const float2*)down)[i];
  float2 r; r.x = d.x + o; r.y = d.y + o;
  ((float2*)out)[i] = r;
}

extern "C" void kernel_launch(void* const* d_in, const int* in_sizes, int n_in,
                              void* d_out, int out_size, void* d_ws, size_t ws_size,
                              hipStream_t stream)
{
  const float* up     = (const float*)d_in[0];
  const float* down   = (const float*)d_in[1];
  const float* vk_w   = (const float*)d_in[2];
  const float* vk_b   = (const float*)d_in[3];
  const float* q_w    = (const float*)d_in[4];
  const float* q_b    = (const float*)d_in[5];
  const float* psi1_w = (const float*)d_in[6];
  const float* psi1_b = (const float*)d_in[7];
  const float* psi2_w = (const float*)d_in[8];
  const float* psi2_b = (const float*)d_in[9];
  const float* phi_w  = (const float*)d_in[10];
  const float* phi_b  = (const float*)d_in[11];

  const int B = 8, C = 256, LUP = 16384, LDOWN = 32768;
  const long long PU = (long long)C * LUP;      // 4,194,304 (per-batch up-size)
  const long long PD = (long long)C * LDOWN;    // 8,388,608
  const long long SU = B * PU;                  // 33,554,432

  // workspace carve (bf16 T-layout intermediates + f32 Ot/S)
  char* p = (char*)d_ws;
  unsigned short* bufA = (unsigned short*)p; p += (size_t)B * PD * 2;       // tD1T / VKdT
  unsigned short* bufB = (unsigned short*)p; p += (size_t)B * PU * 2;       // d2uT/tUT/QuT/QdFT
  unsigned short* bufC = (unsigned short*)p; p += (size_t)B * PD * 2;       // VKuT
  float* Ot = (float*)p;  p += (size_t)SU * 4;
  float* S1 = (float*)p;  p += (size_t)B * C * C * 4;
  float* S2 = (float*)p;

  dim3 blk(256);
  (void)hipMemsetAsync(S1, 0, sizeof(float) * (size_t)B * C * C * 2, stream);

  // 1. tD1T[l][c] = (psi1*down + psi1_b)^T              (bf16, 256ch x 32768)
  gemm_k256<0,1,false,false><<<dim3(LDOWN/128, C/128, B), blk, 0, stream>>>(
      psi1_w, C, 1, 0, down, PD, LDOWN, 0, LDOWN,
      psi1_b, 1.f, nullptr, 0, nullptr, 0, bufA, PD, C, LDOWN);

  // 2. d2uT = (maxpool3_s2(down) + conv3_s2(tD1,psi2) + psi2_b)^T
  gemm_k256<2,1,false,true><<<dim3(LUP/128, C/128, B), blk, 0, stream>>>(
      psi2_w, 3*C, 3, 0, bufA, PD, C, 0, LDOWN,
      psi2_b, 1.f, nullptr, 0, down, PD, bufB, PU, C, LUP);

  // 3. VKdT = (vk*d2u + vk_b)^T                         (512ch x 16384)
  gemm_k256<3,1,false,false><<<dim3(LUP/128, 2*C/128, B), blk, 0, stream>>>(
      vk_w, C, 1, 0, bufB, PU, C, 0, LUP,
      vk_b, 1.f, nullptr, 0, nullptr, 0, bufA, PD, 2*C, LUP);

  // 4. tUT = (up + phi*up + phi_b)^T
  gemm_k256<0,1,true,false><<<dim3(LUP/128, C/128, B), blk, 0, stream>>>(
      phi_w, C, 1, 0, up, PU, LUP, 0, LUP,
      phi_b, 1.f, up, PU, nullptr, 0, bufB, PU, C, LUP);

  // 5. VKuT = (vk*tU + vk_b)^T
  gemm_k256<3,1,false,false><<<dim3(LUP/128, 2*C/128, B), blk, 0, stream>>>(
      vk_w, C, 1, 0, bufB, PU, C, 0, LUP,
      vk_b, 1.f, nullptr, 0, nullptr, 0, bufC, PD, 2*C, LUP);

  // 6. QuT = (q*up + q_b)^T                             (tU dead)
  gemm_k256<0,1,false,false><<<dim3(LUP/128, C/128, B), blk, 0, stream>>>(
      q_w, C, 1, 0, up, PU, LUP, 0, LUP,
      q_b, 1.f, nullptr, 0, nullptr, 0, bufB, PU, C, LUP);

  // 7. S1 = softmax(Qu . K_d2u^T / 16)   (K = channels 256..511 of VKdT)
  gemm_nt_atomic<<<dim3(16, 8, B), blk, 0, stream>>>(
      bufB, PU, C, bufA, PD, 2*C, C, 1.f/16.f, S1);
  softmax_rows<<<dim3(B*C/8), blk, 0, stream>>>(S1, B*C);

  // 8. QdFT = (q*fold2(down) + 2*q_b)^T                 (Qu dead)
  gemm_k256<1,1,false,false><<<dim3(LUP/128, C/128, B), blk, 0, stream>>>(
      q_w, C, 1, 0, down, PD, LDOWN, 0, LDOWN,
      q_b, 2.f, nullptr, 0, nullptr, 0, bufB, PU, C, LUP);

  // 9. S2 = softmax(QdF . k_t^T / 16)    (k = channels 256..511 of VKuT)
  gemm_nt_atomic<<<dim3(16, 8, B), blk, 0, stream>>>(
      bufB, PU, C, bufC, PD, 2*C, C, 1.f/16.f, S2);
  softmax_rows<<<dim3(B*C/8), blk, 0, stream>>>(S2, B*C);

  float* up_out   = (float*)d_out;
  float* down_out = up_out + SU;

  // 10. up_out = up + S1 . V_d2u          (V = channels 0..255 of VKdT; f32 N out)
  gemm_k256<3,0,true,false><<<dim3(LUP/128, C/128, B), blk, 0, stream>>>(
      S1, C, 1, (long long)C*C, bufA, PD, 2*C, 0, LUP,
      nullptr, 0.f, up, PU, nullptr, 0, up_out, PU, 0, LUP);

  // 11. Ot = S2 . v_t                     (v = channels 0..255 of VKuT; f32 N out)
  gemm_k256<3,0,false,false><<<dim3(LUP/128, C/128, B), blk, 0, stream>>>(
      S2, C, 1, (long long)C*C, bufC, PD, 2*C, 0, LUP,
      nullptr, 0.f, nullptr, 0, nullptr, 0, Ot, PU, 0, LUP);

  // 12. down_out = down + duplicate(Ot)
  dup_add<<<dim3((unsigned)(SU/256)), blk, 0, stream>>>(down, Ot, down_out, SU);

  (void)in_sizes; (void)n_in; (void)out_size; (void)ws_size;
}
